// TRN_11639361372798
// MI455X (gfx1250) — compile-verified, matrix operand order checked by reading, not run on hardware
//
#include <hip/hip_runtime.h>
#include <stdint.h>

// ---------------- problem constants ----------------
#define T_ALL   2048
#define NDEC    3
#define T_STEPS (T_ALL - NDEC)   // 2045
#define NWG     16
#define TPB     256              // 8 wave32 per workgroup, 128 waves total

typedef __attribute__((ext_vector_type(16))) __bf16        v16bf;
typedef __attribute__((ext_vector_type(8)))  float         v8f;
typedef __attribute__((ext_vector_type(4)))  unsigned int  v4u;
typedef __attribute__((ext_vector_type(4)))  int           v4i;

union ABfrag { v4u q[2]; unsigned int u[8]; v16bf v; };

// -------- optional async global->LDS staging (ASYNCcnt path) --------
#if defined(__has_builtin)
# if __has_builtin(__builtin_amdgcn_global_load_async_to_lds_b128) && \
     __has_builtin(__builtin_amdgcn_s_wait_asynccnt)
#  define ASYNC_LDS 1
# endif
#endif
#ifndef ASYNC_LDS
# define ASYNC_LDS 0
#endif

__device__ __forceinline__ unsigned short f2bf(float f) {
  unsigned int u = __float_as_uint(f);
  u += 0x7FFFu + ((u >> 16) & 1u);          // round-to-nearest-even
  return (unsigned short)(u >> 16);
}
__device__ __forceinline__ float sigm(float x) { return 1.0f / (1.0f + __expf(-x)); }

// Copy n bf16 (n multiple of 8, 16B-aligned segments) global -> LDS.
__device__ __forceinline__ void stage_bf(const unsigned short* __restrict__ g,
                                         unsigned short* l, int n, int tid) {
#if ASYNC_LDS
  for (int i = tid * 8; i < n; i += TPB * 8)
    __builtin_amdgcn_global_load_async_to_lds_b128(
        (__attribute__((address_space(1))) v4i*)(g + i),
        (__attribute__((address_space(3))) v4i*)(l + i), 0, 0);
#else
  for (int i = tid * 8; i < n; i += TPB * 8)
    *(v4u*)(l + i) = *(const v4u*)(g + i);
#endif
}
__device__ __forceinline__ void stage_wait() {
#if ASYNC_LDS
  __builtin_amdgcn_s_wait_asynccnt(0);
#endif
}

// A fragment: row 0 of a 16x32 bf16 matrix = vector chunk; rows 1..15 zero.
// Non-row-0 lanes read from a zeroed LDS pad (no VALU masking needed).
__device__ __forceinline__ void a_frag(const unsigned short* abase, int koff, int hi,
                                       ABfrag& a) {
#pragma unroll
  for (int v = 0; v < 8; ++v) {
    const int bk = koff + hi * 8 + (v < 4 ? 2 * v : 16 + 2 * (v - 4));
    a.u[v] = *(const unsigned int*)(abase + bk);       // -> 2x ds_load_b128 total
  }
}

// 4 output tiles (r0 + g*rs) sharing one input vector; K-chunks [kc0, kc0+kn).
// 4 independent WMMA accumulator chains, one A fragment per K-chunk.
__device__ __forceinline__ void gemv_tile16_x4(const unsigned short* __restrict__ Wp,
    int r0, int rs, int nKc, int kc0, int kn,
    const unsigned short* sv, const unsigned short* zpad, int lane, float out[4]) {
  const int m = lane & 15, hi = lane >> 4;
  const unsigned short* abase = (m == 0) ? sv : zpad;
  int koff = (m == 0) ? kc0 * 32 : 0;
  const int kstep = (m == 0) ? 32 : 0;
  const unsigned short* bp0 = Wp + ((size_t)(r0)          * nKc + kc0) * 512 + lane * 16;
  const unsigned short* bp1 = Wp + ((size_t)(r0 + rs)     * nKc + kc0) * 512 + lane * 16;
  const unsigned short* bp2 = Wp + ((size_t)(r0 + 2 * rs) * nKc + kc0) * 512 + lane * 16;
  const unsigned short* bp3 = Wp + ((size_t)(r0 + 3 * rs) * nKc + kc0) * 512 + lane * 16;
  v8f c0 = {0.f,0.f,0.f,0.f,0.f,0.f,0.f,0.f};
  v8f c1 = c0, c2 = c0, c3 = c0;
  for (int i = 0; i < kn; ++i) {
    const size_t o = (size_t)i * 512;
    ABfrag a, b0, b1, b2, b3;
    a_frag(abase, koff, hi, a);
    __builtin_prefetch(bp0 + o + 512, 0, 1);
    __builtin_prefetch(bp2 + o + 512, 0, 1);
    b0.q[0] = *(const v4u*)(bp0 + o); b0.q[1] = *(const v4u*)(bp0 + o + 8);
    b1.q[0] = *(const v4u*)(bp1 + o); b1.q[1] = *(const v4u*)(bp1 + o + 8);
    b2.q[0] = *(const v4u*)(bp2 + o); b2.q[1] = *(const v4u*)(bp2 + o + 8);
    b3.q[0] = *(const v4u*)(bp3 + o); b3.q[1] = *(const v4u*)(bp3 + o + 8);
    c0 = __builtin_amdgcn_wmma_f32_16x16x32_bf16(false, a.v, false, b0.v, (short)0, c0, false, false);
    c1 = __builtin_amdgcn_wmma_f32_16x16x32_bf16(false, a.v, false, b1.v, (short)0, c1, false, false);
    c2 = __builtin_amdgcn_wmma_f32_16x16x32_bf16(false, a.v, false, b2.v, (short)0, c2, false, false);
    c3 = __builtin_amdgcn_wmma_f32_16x16x32_bf16(false, a.v, false, b3.v, (short)0, c3, false, false);
    koff += kstep;
  }
  out[0] = c0[0]; out[1] = c1[0]; out[2] = c2[0]; out[3] = c3[0];
}

// Single tile, full K, two interleaved accumulator chains.
__device__ __forceinline__ float gemv_tile16(const unsigned short* __restrict__ Wp,
    int rTile, int nKc, const unsigned short* sv, const unsigned short* zpad, int lane) {
  const int m = lane & 15, hi = lane >> 4;
  const unsigned short* abase = (m == 0) ? sv : zpad;
  int koff = 0;
  const int kstep = (m == 0) ? 32 : 0;
  const unsigned short* bp = Wp + (size_t)rTile * nKc * 512 + lane * 16;
  v8f c0 = {0.f,0.f,0.f,0.f,0.f,0.f,0.f,0.f};
  v8f c1 = c0;
  for (int kc = 0; kc < nKc; kc += 2) {
    const size_t o = (size_t)kc * 512;
    ABfrag a0, a1, b0, b1;
    a_frag(abase, koff, hi, a0);
    a_frag(abase, koff + kstep, hi, a1);
    __builtin_prefetch(bp + o + 1024, 0, 1);
    b0.q[0] = *(const v4u*)(bp + o);       b0.q[1] = *(const v4u*)(bp + o + 8);
    b1.q[0] = *(const v4u*)(bp + o + 512); b1.q[1] = *(const v4u*)(bp + o + 520);
    c0 = __builtin_amdgcn_wmma_f32_16x16x32_bf16(false, a0.v, false, b0.v, (short)0, c0, false, false);
    c1 = __builtin_amdgcn_wmma_f32_16x16x32_bf16(false, a1.v, false, b1.v, (short)0, c1, false, false);
    koff += 2 * kstep;
  }
  return c0[0] + c1[0];
}

// ---------------- grid-wide barrier (graph-capture safe) ----------------
__device__ __forceinline__ void grid_sync(int* cnt, volatile int* gen) {
  __threadfence();
  __syncthreads();
  __builtin_amdgcn_s_cluster_barrier();     // HW fast path if clustered; NOP otherwise
  if (threadIdx.x == 0) {
    int g = *gen;
    if (atomicAdd(cnt, 1) == NWG - 1) {
      atomicExch(cnt, 0);
      __threadfence();
      atomicAdd((int*)gen, 1);
    } else {
      while (*gen == g) __builtin_amdgcn_s_sleep(8);
    }
  }
  __syncthreads();
  __threadfence();
}

// ---------------- weight pack: fp32 row-major -> bf16 WMMA-B tiles ----------------
__global__ void pack_w(const float* __restrict__ W, unsigned short* __restrict__ dst,
                       int Rtiles, int KsrcKc, int kcStart, int KcatKc) {
  int idx = blockIdx.x * blockDim.x + threadIdx.x;
  int total = Rtiles * KsrcKc * 512;
  if (idx >= total) return;
  int pos  = idx & 511;
  int rest = idx >> 9;
  int kc = rest % KsrcKc;
  int rT = rest / KsrcKc;
  int L = pos >> 4;
  int v = (pos >> 1) & 7;
  int e = pos & 1;
  int n = L & 15, hi = L >> 4;
  int kl = (v < 4 ? 2 * v + e : 16 + 2 * (v - 4) + e) + hi * 8;
  int Ksrc = KsrcKc * 32;
  float w = W[(size_t)(rT * 16 + n) * Ksrc + kc * 32 + kl];
  dst[((size_t)rT * KcatKc + (kcStart + kc)) * 512 + pos] = f2bf(w);
}

__global__ void pack_x(const float* __restrict__ X, unsigned short* __restrict__ xb, int n) {
  int i = blockIdx.x * TPB + threadIdx.x;
  if (i < n) xb[i] = f2bf(X[i]);
}

__global__ void init_ws_kernel(int* syncArea, float* fstate, unsigned short* mirrors) {
  int i = blockIdx.x * TPB + threadIdx.x;
  if (i < 16) syncArea[i] = 0;
  if (i < 2048) fstate[i] = 0.f;       // c_enc, dcx
  if (i < 4160) mirrors[i] = 0;        // hb, cb, db, fb, futb
}

// ---------------- persistent LSTM enc/dec kernel ----------------
struct KParams {
  const float *enc_bih, *enc_bhh, *dec_bih, *dec_bhh;
  const float *hx_b, *cx_b, *fus_b, *fut_b, *cls_b;
  const unsigned short *encP, *decP, *hxP, *cxP, *clsP, *fusP, *futP;
  const unsigned short *xb;                 // bf16 input sequence
  unsigned short *hb, *cb, *db, *fb, *futb; // bf16 state mirrors (GEMV inputs)
  float *c_enc, *dcx;                       // fp32 elementwise state
  float *out_enc, *out_dec;
  int *cnt;
  volatile int *gen;
};

__global__ __launch_bounds__(TPB, 1) void lstm_main(KParams p) {
  __shared__ __align__(16) unsigned short svec[2048];  // bf16 input vector(s)
  __shared__ __align__(16) unsigned short sds[64];     // bf16 decoder score (per-WGP)
  __shared__ __align__(16) unsigned short zpad[32];    // zero pad for A rows 1..15
  __shared__ __align__(16) float sgp[256];             // gate partials (K-split pairs)

  const int tid    = threadIdx.x;
  const int lane   = tid & 31;
  const int lw     = tid >> 5;              // wave in workgroup 0..7
  const int gw     = blockIdx.x * 8 + lw;   // global wave 0..127
  const int m      = lane & 15;
  const int pairId = lw & 3;                // K-split pair within workgroup
  const int half   = lw >> 2;               // 0 = low K half, 1 = high K half
  const int s      = blockIdx.x * 4 + pairId; // hidden slice 0..63

  if (tid < 32) zpad[tid] = 0;

  for (int t = 0; t < T_STEPS; ++t) {
    // ---------- Phase E: encoder gates + (h,c) update ----------
    // svec = [ x_t(64) | fut(64) | h(1024) ]  (Kcat = 1152 = 36 chunks)
    stage_bf(p.xb + (size_t)t * 64, svec, 64, tid);
    stage_bf(p.futb, svec + 64, 64, tid);
    stage_bf(p.hb, svec + 128, 1024, tid);
    stage_wait();
    __syncthreads();
    {
      float gv[4];
      gemv_tile16_x4(p.encP, s, 64, 36, half * 18, 18, svec, zpad, lane, gv);
      if (half == 1 && lane < 16) {
#pragma unroll
        for (int g = 0; g < 4; ++g) sgp[pairId * 64 + g * 16 + m] = gv[g];
      }
      __syncthreads();
      if (half == 0) {
#pragma unroll
        for (int g = 0; g < 4; ++g) {       // gate order: i, f, g, o
          int r = g * 64 + s;
          gv[g] += sgp[pairId * 64 + g * 16 + m]
                 + p.enc_bih[r * 16 + m] + p.enc_bhh[r * 16 + m];
        }
        int idx = s * 16 + m;
        float c_old = p.c_enc[idx];
        float cn = sigm(gv[1]) * c_old + sigm(gv[0]) * tanhf(gv[2]);
        float hn = sigm(gv[3]) * tanhf(cn);
        if (lane < 16) {
          p.c_enc[idx] = cn;
          p.hb[idx] = f2bf(hn);
          p.cb[idx] = f2bf(cn);
        }
      }
    }
    grid_sync(p.cnt, p.gen);

    // ---------- Phase H: enc_score, dhx = relu(hx·h), dcx = relu(cx·c) ----------
    stage_bf(p.hb, svec, 1024, tid);
    stage_bf(p.cb, svec + 1024, 1024, tid);
    stage_wait();
    __syncthreads();
    for (int j = gw; j < 132; j += 128) {
      if (j < 4) {
        float v = gemv_tile16(p.clsP, j, 32, svec, zpad, lane) + p.cls_b[j * 16 + m];
        if (lane < 16) p.out_enc[(size_t)t * 64 + j * 16 + m] = v;
      } else if (j < 68) {
        int tt = j - 4;
        float v = gemv_tile16(p.hxP, tt, 32, svec, zpad, lane) + p.hx_b[tt * 16 + m];
        if (lane < 16) p.db[tt * 16 + m] = f2bf(fmaxf(v, 0.f));
      } else {
        int tt = j - 68;
        float v = gemv_tile16(p.cxP, tt, 32, svec + 1024, zpad, lane) + p.cx_b[tt * 16 + m];
        if (lane < 16) p.dcx[tt * 16 + m] = fmaxf(v, 0.f);
      }
    }
    grid_sync(p.cnt, p.gen);

    // ---------- Decoder: 3 sequential iterations (dcx frozen: reference bug) ----------
    float futAcc = 0.f;                     // lives in WG0 waves 4..7
    for (int d = 0; d < NDEC; ++d) {
      // D1: gates = dec_Wih·fusion_in + dec_Whh·dhx ; update dhx only
      if (d == 0) { for (int i = tid; i < 1024; i += TPB) svec[i] = 0; }
      else        { stage_bf(p.fb, svec, 1024, tid); }
      stage_bf(p.db, svec + 1024, 1024, tid);
      stage_wait();
      __syncthreads();
      {
        float gv[4];
        gemv_tile16_x4(p.decP, s, 64, 64, half * 32, 32, svec, zpad, lane, gv);
        if (half == 1 && lane < 16) {
#pragma unroll
          for (int g = 0; g < 4; ++g) sgp[pairId * 64 + g * 16 + m] = gv[g];
        }
        __syncthreads();
        if (half == 0) {
#pragma unroll
          for (int g = 0; g < 4; ++g) {
            int r = g * 64 + s;
            gv[g] += sgp[pairId * 64 + g * 16 + m]
                   + p.dec_bih[r * 16 + m] + p.dec_bhh[r * 16 + m];
          }
          int idx = s * 16 + m;
          float ct = sigm(gv[1]) * p.dcx[idx] + sigm(gv[0]) * tanhf(gv[2]);
          float hn = sigm(gv[3]) * tanhf(ct);
          if (lane < 16) p.db[idx] = f2bf(hn);   // dcx deliberately NOT updated
        }
      }
      grid_sync(p.cnt, p.gen);

      // D2: ds = cls(dhx) (redundant per WGP -> LDS), fut accum, fusion_in = relu(fus·ds)
      stage_bf(p.db, svec, 1024, tid);
      stage_wait();
      __syncthreads();
      if (lw < 4) {                         // every WGP: decoder score (4 tiles)
        float v = gemv_tile16(p.clsP, lw, 32, svec, zpad, lane) + p.cls_b[lw * 16 + m];
        if (lane < 16) {
          sds[lw * 16 + m] = f2bf(v);
          if (blockIdx.x == 0) p.out_dec[((size_t)t * NDEC + d) * 64 + lw * 16 + m] = v;
        }
      } else if (blockIdx.x == 0) {         // WG0 waves 4..7: fut head (4 tiles)
        int tt = lw - 4;
        float v = gemv_tile16(p.futP, tt, 32, svec, zpad, lane) + p.fut_b[tt * 16 + m];
        futAcc += fmaxf(v, 0.f);
        if (d == NDEC - 1 && lane < 16) p.futb[tt * 16 + m] = f2bf(futAcc * (1.f / 3.f));
      }
      __syncthreads();                      // sds ready (WGP-local sync only)
      for (int j = gw; j < 64; j += 128) {  // fusion_in = relu(fus_W·ds + b), K = 64
        float v = gemv_tile16(p.fusP, j, 2, sds, zpad, lane) + p.fus_b[j * 16 + m];
        if (lane < 16) p.fb[j * 16 + m] = f2bf(fmaxf(v, 0.f));
      }
      grid_sync(p.cnt, p.gen);
    }
  }
}

// ---------------- host-side launch ----------------
extern "C" void kernel_launch(void* const* d_in, const int* in_sizes, int n_in,
                              void* d_out, int out_size, void* d_ws, size_t ws_size,
                              hipStream_t stream) {
  (void)in_sizes; (void)n_in; (void)out_size; (void)ws_size;
  const float* X       = (const float*)d_in[0];
  const float* enc_Wih = (const float*)d_in[1];
  const float* enc_Whh = (const float*)d_in[2];
  const float* enc_bih = (const float*)d_in[3];
  const float* enc_bhh = (const float*)d_in[4];
  const float* dec_Wih = (const float*)d_in[5];
  const float* dec_Whh = (const float*)d_in[6];
  const float* dec_bih = (const float*)d_in[7];
  const float* dec_bhh = (const float*)d_in[8];
  const float* hx_W    = (const float*)d_in[9];
  const float* hx_b    = (const float*)d_in[10];
  const float* cx_W    = (const float*)d_in[11];
  const float* cx_b    = (const float*)d_in[12];
  const float* fus_W   = (const float*)d_in[13];
  const float* fus_b   = (const float*)d_in[14];
  const float* fut_W   = (const float*)d_in[15];
  const float* fut_b   = (const float*)d_in[16];
  const float* cls_W   = (const float*)d_in[17];
  const float* cls_b   = (const float*)d_in[18];

  // workspace layout
  char* ws = (char*)d_ws;
  int*   syncArea = (int*)ws;                               // 64 B
  float* fstate   = (float*)(ws + 64);                      // c_enc(1024), dcx(1024)
  float* c_enc = fstate;
  float* dcx   = fstate + 1024;
  unsigned short* mirrors = (unsigned short*)(ws + 64 + 2048 * sizeof(float)); // 8256
  unsigned short* hb   = mirrors;           // 1024
  unsigned short* cb   = hb + 1024;
  unsigned short* db   = cb + 1024;
  unsigned short* fb   = db + 1024;
  unsigned short* futb = fb + 1024;         // 64
  unsigned short* xb   = mirrors + 4160;    // T_ALL*64 = 131072  (offset 16576, 64B aligned)
  unsigned short* packBase = xb + (size_t)T_ALL * 64;       // offset 278720, 64B aligned
  unsigned short* encP = packBase;                          // 256*36*512
  unsigned short* decP = encP + (size_t)256 * 36 * 512;     // 256*64*512
  unsigned short* hxP  = decP + (size_t)256 * 64 * 512;     // 64*32*512
  unsigned short* cxP  = hxP  + (size_t)64 * 32 * 512;
  unsigned short* clsP = cxP  + (size_t)64 * 32 * 512;      // 4*32*512
  unsigned short* fusP = clsP + (size_t)4 * 32 * 512;       // 64*2*512
  unsigned short* futP = fusP + (size_t)64 * 2 * 512;       // 4*32*512

  init_ws_kernel<<<(4160 + TPB - 1) / TPB, TPB, 0, stream>>>(syncArea, fstate, mirrors);
  pack_x<<<(T_ALL * 64 + TPB - 1) / TPB, TPB, 0, stream>>>(X, xb, T_ALL * 64);

  auto launch_pack = [&](const float* W, unsigned short* dst,
                         int Rtiles, int KsrcKc, int kcStart, int KcatKc) {
    int total = Rtiles * KsrcKc * 512;
    pack_w<<<(total + TPB - 1) / TPB, TPB, 0, stream>>>(W, dst, Rtiles, KsrcKc, kcStart, KcatKc);
  };
  // encoder concat: [Wih (K=128) | Whh (K=1024)] -> Kcat = 1152 (36 chunks)
  launch_pack(enc_Wih, encP, 256, 4, 0, 36);
  launch_pack(enc_Whh, encP, 256, 32, 4, 36);
  // decoder concat: [Wih (K=1024) | Whh (K=1024)] -> Kcat = 2048 (64 chunks)
  launch_pack(dec_Wih, decP, 256, 32, 0, 64);
  launch_pack(dec_Whh, decP, 256, 32, 32, 64);
  launch_pack(hx_W,  hxP,  64, 32, 0, 32);
  launch_pack(cx_W,  cxP,  64, 32, 0, 32);
  launch_pack(cls_W, clsP, 4, 32, 0, 32);
  launch_pack(fus_W, fusP, 64, 2, 0, 2);
  launch_pack(fut_W, futP, 4, 32, 0, 32);

  KParams p;
  p.enc_bih = enc_bih; p.enc_bhh = enc_bhh;
  p.dec_bih = dec_bih; p.dec_bhh = dec_bhh;
  p.hx_b = hx_b; p.cx_b = cx_b; p.fus_b = fus_b; p.fut_b = fut_b; p.cls_b = cls_b;
  p.encP = encP; p.decP = decP; p.hxP = hxP; p.cxP = cxP;
  p.clsP = clsP; p.fusP = fusP; p.futP = futP;
  p.xb = xb;
  p.hb = hb; p.cb = cb; p.db = db; p.fb = fb; p.futb = futb;
  p.c_enc = c_enc; p.dcx = dcx;
  p.out_enc = (float*)d_out;
  p.out_dec = (float*)d_out + (size_t)T_STEPS * 64;
  p.cnt = syncArea;
  p.gen = (volatile int*)(syncArea + 1);

  lstm_main<<<NWG, TPB, 0, stream>>>(p);
}
